// Attention_43911745634710
// MI455X (gfx1250) — matrix operand order repaired and checked by reading, not compiled, not run on hardware
//
#include <hip/hip_runtime.h>

// ---------------------------------------------------------------------------
// GQA prefill attention for gfx1250 (MI455X): QKV proj + RoPE + causal flash
// attention + O proj. All matmuls via v_wmma_f32_16x16x32_bf16 (wave32).
// GEMMs are 32x64 register-blocked; __launch_bounds__(.,1) prevents the
// occupancy-driven VGPR cap from spilling resident fragments/accumulators.
// ---------------------------------------------------------------------------

typedef __attribute__((ext_vector_type(16))) __bf16 v16bf;
typedef __attribute__((ext_vector_type(8)))  __bf16 v8bf;
typedef __attribute__((ext_vector_type(8)))  float  v8f;

namespace {
constexpr int Bb = 2, Tt = 2048, Dd = 2048;
constexpr int Nn = 16, Kk = 4, Hh = 128;
constexpr int Gg = Nn / Kk;          // 4
constexpr int BT = Bb * Tt;          // 4096
constexpr int NH = Nn * Hh;          // 2048
constexpr int KH = Kk * Hh;          // 512
constexpr int TPB = 256;             // 8 waves per block
}

// ---- bf16 helpers (explicit RNE, storage-only use of __bf16) --------------
__device__ __forceinline__ __bf16 f2bf(float f) {
  union { unsigned short u; __bf16 b; } cv;
  unsigned u = __float_as_uint(f);
  cv.u = (unsigned short)((u + 0x7FFFu + ((u >> 16) & 1u)) >> 16);
  return cv.b;
}
__device__ __forceinline__ float bf2f(__bf16 b) {
  union { __bf16 b; unsigned short u; } cv; cv.b = b;
  return __uint_as_float(((unsigned)cv.u) << 16);
}

// ---- WMMA fragment loaders (contiguous k = 16h + s order, same for A and B) -----------------
// A (16xK32 bf16): lane<16 -> row=lane, K k0..k0+15
//                  lane>=16 -> row=lane-16, K k0+16..k0+31
__device__ __forceinline__ v16bf load_a_frag(const __bf16* base, int ld, int lane, int k0) {
  int row = lane & 15;
  int kk  = k0 + ((lane >> 4) << 4);
  const __bf16* p = base + (size_t)row * ld + kk;
  v8bf lo = *(const v8bf*)(p);
  v8bf hi = *(const v8bf*)(p + 8);
  v16bf r;
#pragma unroll
  for (int i = 0; i < 8; ++i) { r[i] = lo[i]; r[8 + i] = hi[i]; }
  return r;
}
// B (K32x16 bf16) from B^T row-major [col, K]: lane<16 -> K 0..15, lane>=16 -> K 16..31
__device__ __forceinline__ v16bf load_b_frag(const __bf16* base, int ld, int lane, int k0) {
  int col = lane & 15;
  int kk  = k0 + ((lane >> 4) << 4);
  const __bf16* p = base + (size_t)col * ld + kk;
  v8bf lo = *(const v8bf*)(p);
  v8bf hi = *(const v8bf*)(p + 8);
  v16bf r;
#pragma unroll
  for (int i = 0; i < 8; ++i) { r[i] = lo[i]; r[8 + i] = hi[i]; }
  return r;
}
__device__ __forceinline__ v8f wmma_bf16(v16bf a, v16bf b, v8f c) {
  return __builtin_amdgcn_wmma_f32_16x16x32_bf16(false, a, false, b, (short)0, c, false, false);
}

// ---- prep kernels ---------------------------------------------------------
__global__ void k_cvt(const float* __restrict__ x, __bf16* __restrict__ y, int n) {
  for (int i = blockIdx.x * blockDim.x + threadIdx.x; i < n; i += gridDim.x * blockDim.x)
    y[i] = f2bf(x[i]);
}
// in f32 [rows, cols] -> out bf16 [cols, rows]
__global__ void k_transpose(const float* __restrict__ in, __bf16* __restrict__ out,
                            int rows, int cols) {
  int n = rows * cols;
  for (int i = blockIdx.x * blockDim.x + threadIdx.x; i < n; i += gridDim.x * blockDim.x) {
    int r = i / cols, c = i - r * cols;
    out[(size_t)c * rows + r] = f2bf(in[i]);
  }
}

// ---- GEMM: C[M,Nout] = A[M,Kd] * Bt[Nout,Kd]^T ----------------------------
// One wave computes a 32x64 block: 2 A-frags x 4 B-frags -> 8 WMMA / k-step.
template <typename OutT>
__global__ void __launch_bounds__(TPB, 1)
k_gemm_blk(const __bf16* __restrict__ A, const __bf16* __restrict__ Bt,
           OutT* __restrict__ C, int M, int Nout, int Kd) {
  int lane = threadIdx.x & 31;
  int wave = (int)((blockIdx.x * blockDim.x + threadIdx.x) >> 5);
  int tn_cnt = Nout >> 6;                       // 64-wide N blocks
  int tm = wave / tn_cnt, tn = wave - tm * tn_cnt;
  if ((tm << 5) >= M) return;
  const __bf16* Ab = A  + (size_t)tm * 32 * Kd;
  const __bf16* Bb2 = Bt + (size_t)tn * 64 * Kd;

  v8f acc[2][4];
#pragma unroll
  for (int mi = 0; mi < 2; ++mi)
#pragma unroll
    for (int ni = 0; ni < 4; ++ni)
#pragma unroll
      for (int j = 0; j < 8; ++j) acc[mi][ni][j] = 0.0f;

  for (int k0 = 0; k0 < Kd; k0 += 32) {
    // prefetch next k-chunk one iteration ahead (global_prefetch_b8)
    if (k0 + 32 < Kd) {
      __builtin_prefetch(Ab + (size_t)(lane & 15) * Kd + k0 + 32, 0, 3);
      __builtin_prefetch(Bb2 + (size_t)(lane & 15) * Kd + k0 + 32, 0, 3);
    }
    v16bf a0 = load_a_frag(Ab,                     Kd, lane, k0);
    v16bf a1 = load_a_frag(Ab + (size_t)16 * Kd,   Kd, lane, k0);
#pragma unroll
    for (int ni = 0; ni < 4; ++ni) {
      v16bf b = load_b_frag(Bb2 + (size_t)(ni * 16) * Kd, Kd, lane, k0);
      acc[0][ni] = wmma_bf16(a0, b, acc[0][ni]);
      acc[1][ni] = wmma_bf16(a1, b, acc[1][ni]);
    }
  }

  int colb = (tn << 6) + (lane & 15);
  int r0   = (tm << 5) + ((lane >> 4) << 3);
#pragma unroll
  for (int mi = 0; mi < 2; ++mi)
#pragma unroll
    for (int ni = 0; ni < 4; ++ni)
#pragma unroll
      for (int j = 0; j < 8; ++j) {
        size_t idx = (size_t)(r0 + mi * 16 + j) * Nout + colb + ni * 16;
        if constexpr (sizeof(OutT) == 4) C[idx] = acc[mi][ni][j];
        else                             C[idx] = f2bf(acc[mi][ni][j]);
      }
}

// ---- RoPE in place on [BT, heads, H] bf16 ---------------------------------
__global__ void k_rope(__bf16* __restrict__ buf, const int* __restrict__ pos, int heads) {
  constexpr int half = Hh / 2;  // 64
  int n = BT * heads * half;
  for (int idx = blockIdx.x * blockDim.x + threadIdx.x; idx < n; idx += gridDim.x * blockDim.x) {
    int h   = idx % half;
    int tmp = idx / half;
    int hd  = tmp % heads;
    int bt  = tmp / heads;               // b*T + t
    float p = (float)pos[bt];
    float ts = __powf(10000.0f, (float)h / (float)half);  // MIN_TS*(MAX/MIN)^(h/half)
    float ang = p / ts;
    float s, c; __sincosf(ang, &s, &c);
    __bf16* row = buf + ((size_t)bt * heads + hd) * Hh;
    float x1 = bf2f(row[h]), x2 = bf2f(row[h + half]);
    row[h]        = f2bf(x1 * c - x2 * s);
    row[h + half] = f2bf(x2 * c + x1 * s);
  }
}

// ---- V transpose: Vr[b,s,k,h] -> VT[b,k,h,s] ------------------------------
__global__ void k_vt(const __bf16* __restrict__ Vr, __bf16* __restrict__ VT) {
  int n = BT * KH;
  for (int i = blockIdx.x * blockDim.x + threadIdx.x; i < n; i += gridDim.x * blockDim.x) {
    int h   = i % Hh;
    int tmp = i / Hh;
    int k   = tmp % Kk;
    int bt  = tmp / Kk;
    int s   = bt % Tt;
    int b   = bt / Tt;
    VT[(((size_t)b * Kk + k) * Hh + h) * Tt + s] = Vr[i];
  }
}

// ---- flash attention: one wave per (b, head, 16-query tile) ---------------
__global__ void __launch_bounds__(TPB, 1)
k_flash(const __bf16* __restrict__ Q, const __bf16* __restrict__ Kc,
        const __bf16* __restrict__ VT, __bf16* __restrict__ O) {
  __shared__ __align__(16) __bf16 plds[TPB / 32][16 * 32];
  int lane = threadIdx.x & 31;
  int wib  = threadIdx.x >> 5;
  int wave = blockIdx.x * (blockDim.x >> 5) + wib;
  constexpr int QT = Tt / 16;  // 128
  if (wave >= Bb * Nn * QT) return;
  int b   = wave / (Nn * QT);
  int rem = wave - b * (Nn * QT);
  int n   = rem / QT;
  int qt  = rem - n * QT;
  int q0  = qt << 4;
  int kvh = n / Gg;  // query head n -> kv head n/G

  const __bf16* Qrow = Q + ((size_t)(b * Tt + q0) * NH) + n * Hh;   // ld = NH
  const __bf16* Krow = Kc + (size_t)b * Tt * KH + kvh * Hh;         // ld = KH (row = key s)
  const __bf16* Vth  = VT + ((size_t)(b * Kk + kvh) * Hh) * Tt;     // [h, s], ld = Tt
  __bf16* myl = &plds[wib][0];

  // resident Q fragments covering H=128 (4 x K32)
  v16bf qa[4];
#pragma unroll
  for (int i = 0; i < 4; ++i) qa[i] = load_a_frag(Qrow, NH, lane, i * 32);

  v8f acc[8];
#pragma unroll
  for (int t = 0; t < 8; ++t)
#pragma unroll
    for (int j = 0; j < 8; ++j) acc[t][j] = 0.0f;

  float m_i[8], l_i[8];
#pragma unroll
  for (int j = 0; j < 8; ++j) { m_i[j] = -1e30f; l_i[j] = 0.0f; }

  int rbase = (lane >> 4) << 3;   // lanes 0-15 own rows 0-7, lanes 16-31 rows 8-15
  int colid = lane & 15;

  for (int s0 = 0; s0 <= q0 + 15; s0 += 32) {
    // scores: two 16x16 tiles (keys s0..+15 and s0+16..+31), K-loop over H
    v8f sc0 = {}, sc1 = {};
    const __bf16* kb0 = Krow + (size_t)s0 * KH;
    const __bf16* kb1 = Krow + (size_t)(s0 + 16) * KH;
    if (s0 + 32 <= q0 + 15) {  // prefetch next key chunk
      __builtin_prefetch(kb0 + (size_t)32 * KH + (size_t)colid * KH, 0, 3);
      __builtin_prefetch(kb1 + (size_t)32 * KH + (size_t)colid * KH, 0, 3);
    }
#pragma unroll
    for (int i = 0; i < 4; ++i) {
      sc0 = wmma_bf16(qa[i], load_b_frag(kb0, KH, lane, i * 32), sc0);
      sc1 = wmma_bf16(qa[i], load_b_frag(kb1, KH, lane, i * 32), sc1);
    }
    // online softmax (scale == 1.0 per reference)
    float alpha[8];
#pragma unroll
    for (int j = 0; j < 8; ++j) {
      int r = q0 + rbase + j;
      float v0 = (s0 + colid      > r) ? -1e30f : sc0[j];
      float v1 = (s0 + 16 + colid > r) ? -1e30f : sc1[j];
      float mx = fmaxf(v0, v1);
      mx = fmaxf(mx, __shfl_xor(mx, 1));
      mx = fmaxf(mx, __shfl_xor(mx, 2));
      mx = fmaxf(mx, __shfl_xor(mx, 4));
      mx = fmaxf(mx, __shfl_xor(mx, 8));
      float mn = fmaxf(m_i[j], mx);
      float a  = __expf(m_i[j] - mn);
      float p0 = __expf(v0 - mn);
      float p1 = __expf(v1 - mn);
      float rs = p0 + p1;
      rs += __shfl_xor(rs, 1);
      rs += __shfl_xor(rs, 2);
      rs += __shfl_xor(rs, 4);
      rs += __shfl_xor(rs, 8);
      l_i[j] = l_i[j] * a + rs;
      m_i[j] = mn;
      alpha[j] = a;
      // stash P (C layout) into LDS 16x32 tile for A-layout re-read
      myl[(rbase + j) * 32 + colid]      = f2bf(p0);
      myl[(rbase + j) * 32 + 16 + colid] = f2bf(p1);
    }
    v16bf pa = load_a_frag(myl, 32, lane, 0);  // same-wave DS ordering
    // O += P * V  (8 h-tiles), with per-row rescale by alpha
#pragma unroll
    for (int t = 0; t < 8; ++t) {
#pragma unroll
      for (int j = 0; j < 8; ++j) acc[t][j] *= alpha[j];
      v16bf vf = load_b_frag(Vth + (size_t)(t * 16) * Tt + s0, Tt, lane, 0);
      acc[t] = wmma_bf16(pa, vf, acc[t]);
    }
  }

  // epilogue: normalize and store attn [b,t,n,h] bf16
#pragma unroll
  for (int t = 0; t < 8; ++t)
#pragma unroll
    for (int j = 0; j < 8; ++j) {
      float o = acc[t][j] / l_i[j];
      O[(size_t)(b * Tt + q0 + rbase + j) * NH + n * Hh + t * 16 + colid] = f2bf(o);
    }
}

// ---------------------------------------------------------------------------
extern "C" void kernel_launch(void* const* d_in, const int* in_sizes, int n_in,
                              void* d_out, int out_size, void* d_ws, size_t ws_size,
                              hipStream_t stream) {
  const float* Xq   = (const float*)d_in[0];
  const float* Xkv  = (const float*)d_in[1];
  const int*   qpos = (const int*)d_in[2];
  const int*   kpos = (const int*)d_in[3];
  const float* Wq   = (const float*)d_in[4];
  const float* Wk   = (const float*)d_in[5];
  const float* Wv   = (const float*)d_in[6];
  const float* Wo   = (const float*)d_in[7];
  float* out = (float*)d_out;

  char* ws = (char*)d_ws;
  size_t off = 0;
  auto take = [&](size_t elems) {
    __bf16* p = (__bf16*)(ws + off);
    off += ((elems * sizeof(__bf16) + 255) & ~(size_t)255);
    return p;
  };
  __bf16* XqB  = take((size_t)BT * Dd);
  __bf16* XkvB = take((size_t)BT * Dd);
  __bf16* WqT  = take((size_t)NH * Dd);
  __bf16* WkT  = take((size_t)KH * Dd);
  __bf16* WvT  = take((size_t)KH * Dd);
  __bf16* WoT  = take((size_t)Dd * NH);
  __bf16* Qb   = take((size_t)BT * NH);
  __bf16* Kb   = take((size_t)BT * KH);
  __bf16* Vr   = take((size_t)BT * KH);
  __bf16* Vt   = take((size_t)BT * KH);
  __bf16* attn = XqB;  // XqB dead after Q projection; reuse for attention output

  // 1) prep: activation conversion + weight transpose/convert (B^T layout)
  k_cvt<<<2048, TPB, 0, stream>>>(Xq,  XqB,  BT * Dd);
  k_cvt<<<2048, TPB, 0, stream>>>(Xkv, XkvB, BT * Dd);
  k_transpose<<<2048, TPB, 0, stream>>>(Wq, WqT, Dd, NH);   // [D,N*H] -> [N*H,D]
  k_transpose<<<1024, TPB, 0, stream>>>(Wk, WkT, Dd, KH);
  k_transpose<<<1024, TPB, 0, stream>>>(Wv, WvT, Dd, KH);
  k_transpose<<<2048, TPB, 0, stream>>>(Wo, WoT, NH, Dd);   // [N*H,D] -> [D,N*H]

  // 2) projections (32x64 blocked waves)
  auto gemm_blocks = [](int M, int Nout) {
    long waves = (long)(M >> 5) * (Nout >> 6);
    return (int)((waves * 32 + TPB - 1) / TPB);
  };
  k_gemm_blk<__bf16><<<gemm_blocks(BT, NH), TPB, 0, stream>>>(XqB,  WqT, Qb, BT, NH, Dd);
  k_gemm_blk<__bf16><<<gemm_blocks(BT, KH), TPB, 0, stream>>>(XkvB, WkT, Kb, BT, KH, Dd);
  k_gemm_blk<__bf16><<<gemm_blocks(BT, KH), TPB, 0, stream>>>(XkvB, WvT, Vr, BT, KH, Dd);

  // 3) RoPE (in place) + V transpose
  k_rope<<<2048, TPB, 0, stream>>>(Qb, qpos, Nn);
  k_rope<<<1024, TPB, 0, stream>>>(Kb, kpos, Kk);
  k_vt<<<1024, TPB, 0, stream>>>(Vr, Vt);

  // 4) causal flash attention (GQA)
  {
    int waves = Bb * Nn * (Tt / 16);
    int blks  = (waves * 32 + TPB - 1) / TPB;
    k_flash<<<blks, TPB, 0, stream>>>(Qb, Kb, Vt, attn);
  }

  // 5) output projection -> f32 d_out
  k_gemm_blk<float><<<gemm_blocks(BT, Dd), TPB, 0, stream>>>(attn, WoT, out, BT, Dd, NH);
}
